// PAConv_67448166416921
// MI455X (gfx1250) — matrix operand order repaired and checked by reading, not compiled, not run on hardware
//
#include <hip/hip_runtime.h>
#include <hip/hip_bf16.h>

// ---------------------------------------------------------------------------
// PAConv forward for MI455X (gfx1250, wave32, WMMA).
// GEMMs: v_wmma_f32_16x16x32_f16, 16x64 output per wave, K unrolled x2
// (8 WMMAs per iteration, disjoint fragment registers for load/WMMA overlap),
// all operand loads as float4 -> global_load_b128.
// ---------------------------------------------------------------------------

#define EPSBN 1e-5f
#define BB  4
#define NN  2048
#define KNB 20
#define MM  8
#define BN  (BB*NN)       // 8192 points
#define BNK (BN*KNB)      // 163840 (point, neighbor) pairs

typedef __attribute__((ext_vector_type(16))) _Float16 v16h;
typedef __attribute__((ext_vector_type(8)))  float    v8f;

__device__ __forceinline__ void pack8(v16h& v, int base, float4 p0, float4 p1)
{
    v[base + 0] = (_Float16)p0.x; v[base + 1] = (_Float16)p0.y;
    v[base + 2] = (_Float16)p0.z; v[base + 3] = (_Float16)p0.w;
    v[base + 4] = (_Float16)p1.x; v[base + 5] = (_Float16)p1.y;
    v[base + 6] = (_Float16)p1.z; v[base + 7] = (_Float16)p1.w;
}

// A fragment: contiguous K order, element s = K 16*half + s (same order as loadB)
__device__ __forceinline__ v16h loadA(const float* ar, int half)
{
    v16h a;
    const float4* p = (const float4*)(ar + 16 * half);
    const float4* q = (const float4*)(ar + 16 * half + 8);
    pack8(a, 0, p[0], p[1]);
    pack8(a, 8, q[0], q[1]);
    return a;
}

// B fragment: 16 contiguous K values of one W row
__device__ __forceinline__ v16h loadB(const float* wp)
{
    v16h b;
    const float4* p = (const float4*)wp;
    pack8(b, 0, p[0], p[1]);
    pack8(b, 8, p[2], p[3]);
    return b;
}

// ---------------------------------------------------------------------------
// WMMA GEMM: Out[r, c] = sum_k A[r, k] * W[c, k]
//   A: rows x Ktot f32, row-major, leading dim lda (lda % 4 == 0)
//   W: cols x Ktot f32, row-major (out-major), leading dim ldw (ldw % 4 == 0)
// One wave produces a 16x64 tile; K stepped by 64 (2 chunks, 8 v_wmma).
// rows%16==0, Ktot%64==0 required; cols guarded.
// ---------------------------------------------------------------------------
__global__ void gemm_wmma_kernel(const float* __restrict__ A,
                                 const float* __restrict__ W,
                                 float* __restrict__ Out,
                                 int Ktot, int cols, int lda, int ldw, int ldo)
{
    const int lane = threadIdx.x;        // 0..31
    const int m    = lane & 15;
    const int half = lane >> 4;
    const int row0 = blockIdx.x * 16;
    const int col0 = blockIdx.y * 64;

    const int bBase = 16 * half;         // B fragment: K = bBase + j

    // clamped W row per sub-tile (OOB-safe reads; stores guarded)
    const float* wr0; const float* wr1; const float* wr2; const float* wr3;
    {
        int c0 = col0 + 0 * 16 + m;  c0 = (c0 < cols) ? c0 : (cols - 1);
        int c1 = col0 + 1 * 16 + m;  c1 = (c1 < cols) ? c1 : (cols - 1);
        int c2 = col0 + 2 * 16 + m;  c2 = (c2 < cols) ? c2 : (cols - 1);
        int c3 = col0 + 3 * 16 + m;  c3 = (c3 < cols) ? c3 : (cols - 1);
        wr0 = W + (size_t)c0 * ldw + bBase;
        wr1 = W + (size_t)c1 * ldw + bBase;
        wr2 = W + (size_t)c2 * ldw + bBase;
        wr3 = W + (size_t)c3 * ldw + bBase;
    }
    const float* arow = A + (size_t)(row0 + m) * lda;

    v8f acc0 = {}, acc1 = {}, acc2 = {}, acc3 = {};

    for (int k0 = 0; k0 < Ktot; k0 += 64) {
        // chunk 0 fragments
        v16h a0  = loadA(arow + k0, half);
        v16h b00 = loadB(wr0 + k0);
        v16h b01 = loadB(wr1 + k0);
        v16h b02 = loadB(wr2 + k0);
        v16h b03 = loadB(wr3 + k0);
        // chunk 1 fragments (disjoint registers -> overlap with chunk-0 WMMAs)
        v16h a1  = loadA(arow + k0 + 32, half);
        v16h b10 = loadB(wr0 + k0 + 32);
        v16h b11 = loadB(wr1 + k0 + 32);
        v16h b12 = loadB(wr2 + k0 + 32);
        v16h b13 = loadB(wr3 + k0 + 32);

        acc0 = __builtin_amdgcn_wmma_f32_16x16x32_f16(false, a0, false, b00, (short)0, acc0, false, false);
        acc1 = __builtin_amdgcn_wmma_f32_16x16x32_f16(false, a0, false, b01, (short)0, acc1, false, false);
        acc2 = __builtin_amdgcn_wmma_f32_16x16x32_f16(false, a0, false, b02, (short)0, acc2, false, false);
        acc3 = __builtin_amdgcn_wmma_f32_16x16x32_f16(false, a0, false, b03, (short)0, acc3, false, false);
        acc0 = __builtin_amdgcn_wmma_f32_16x16x32_f16(false, a1, false, b10, (short)0, acc0, false, false);
        acc1 = __builtin_amdgcn_wmma_f32_16x16x32_f16(false, a1, false, b11, (short)0, acc1, false, false);
        acc2 = __builtin_amdgcn_wmma_f32_16x16x32_f16(false, a1, false, b12, (short)0, acc2, false, false);
        acc3 = __builtin_amdgcn_wmma_f32_16x16x32_f16(false, a1, false, b13, (short)0, acc3, false, false);
    }

    // C/D layout: VGPR r -> M = r + 8*half, N = lane&15
#pragma unroll
    for (int t = 0; t < 4; ++t) {
        int gc = col0 + t * 16 + m;
        if (gc < cols) {
            v8f acc = (t == 0) ? acc0 : (t == 1) ? acc1 : (t == 2) ? acc2 : acc3;
#pragma unroll
            for (int r = 0; r < 8; ++r) {
                int gr = row0 + r + 8 * half;
                Out[(size_t)gr * ldo + gc] = acc[r];
            }
        }
    }
}

// transpose (R x C) -> (C x R); used once per stage for `mat` (128 x 512)
__global__ void transpose_kernel(const float* __restrict__ in, float* __restrict__ out,
                                 int R, int C)
{
    int i = blockIdx.x * blockDim.x + threadIdx.x;
    if (i >= R * C) return;
    int r = i / C, c = i - r * C;
    out[(size_t)c * R + r] = in[i];
}

// ---------------------------------------------------------------------------
// kNN: per-thread register-resident top-20 (negative squared distance, incl self)
// ---------------------------------------------------------------------------
__global__ void knn_kernel(const float* __restrict__ x, int* __restrict__ idxB)
{
    __shared__ float sx[256], sy[256], sz[256];
    int b = blockIdx.x >> 3;                         // N/256 = 8 blocks per batch
    int n = ((blockIdx.x & 7) << 8) + threadIdx.x;
    const float* xb = x + (size_t)b * 3 * NN;
    float px = xb[n], py = xb[NN + n], pz = xb[2 * NN + n];

    float best[KNB]; int bidx[KNB];
#pragma unroll
    for (int i = 0; i < KNB; ++i) { best[i] = -3.4e38f; bidx[i] = 0; }

    for (int m0 = 0; m0 < NN; m0 += 256) {
        int t = threadIdx.x;
        sx[t] = xb[m0 + t]; sy[t] = xb[NN + m0 + t]; sz[t] = xb[2 * NN + m0 + t];
        __syncthreads();
        for (int jj = 0; jj < 256; ++jj) {
            float dx = px - sx[jj], dy = py - sy[jj], dz = pz - sz[jj];
            float d = -(dx * dx + dy * dy + dz * dz);
            if (d > best[KNB - 1]) {
                best[KNB - 1] = d; bidx[KNB - 1] = m0 + jj;
#pragma unroll
                for (int i = KNB - 1; i > 0; --i) {      // single bubble pass
                    if (best[i] > best[i - 1]) {
                        float tv = best[i]; best[i] = best[i - 1]; best[i - 1] = tv;
                        int   ti = bidx[i]; bidx[i] = bidx[i - 1]; bidx[i - 1] = ti;
                    }
                }
            }
        }
        __syncthreads();
    }
    int* o = idxB + (size_t)(b * NN + n) * KNB;
#pragma unroll
    for (int i = 0; i < KNB; ++i) o[i] = bidx[i];
}

// ---------------------------------------------------------------------------
// xyz10[bnk, 0..9] = [ctr(3), diff(3), nb(3), dist]
// ---------------------------------------------------------------------------
__global__ void xyz10_kernel(const float* __restrict__ x, const int* __restrict__ idxB,
                             float* __restrict__ xyz)
{
    int gid = blockIdx.x * blockDim.x + threadIdx.x;
    if (gid >= BNK) return;
    int bn = gid / KNB;
    int n  = bn % NN;
    int b  = bn / NN;
    const float* xb = x + (size_t)b * 3 * NN;
    int j = idxB[gid];
    float cx = xb[n],  cy = xb[NN + n],  cz = xb[2 * NN + n];
    float nx = xb[j],  ny = xb[NN + j],  nz = xb[2 * NN + j];
    float dx = nx - cx, dy = ny - cy, dz = nz - cz;
    float dist = sqrtf(dx * dx + dy * dy + dz * dz + 1e-12f);
    float* o = xyz + (size_t)gid * 10;
    o[0] = cx; o[1] = cy; o[2] = cz;
    o[3] = dx; o[4] = dy; o[5] = dz;
    o[6] = nx; o[7] = ny; o[8] = nz; o[9] = dist;
}

__global__ void zero_kernel(float* p, int n)
{
    int i = blockIdx.x * blockDim.x + threadIdx.x;
    if (i < n) p[i] = 0.f;
}

// ---------------------------------------------------------------------------
// First + second moments of the 10 xyz channels over all BNK samples.
// mom[0..9] = sum x_i ; mom[10 + tri(i,j)] = sum x_i x_j (i<=j), 55 entries.
// ---------------------------------------------------------------------------
__global__ void moments_kernel(const float* __restrict__ xyz, float* __restrict__ mom)
{
    float acc[65];
#pragma unroll
    for (int i = 0; i < 65; ++i) acc[i] = 0.f;
    int stride = gridDim.x * blockDim.x;
    for (int p = blockIdx.x * blockDim.x + threadIdx.x; p < BNK; p += stride) {
        float v[10];
        const float* src = xyz + (size_t)p * 10;
#pragma unroll
        for (int i = 0; i < 10; ++i) v[i] = src[i];
#pragma unroll
        for (int i = 0; i < 10; ++i) acc[i] += v[i];
#pragma unroll
        for (int i = 0; i < 10; ++i)
#pragma unroll
            for (int j = i; j < 10; ++j)
                acc[10 + i * 10 - (i * (i + 1)) / 2 + j] += v[i] * v[j];
    }
#pragma unroll
    for (int q = 0; q < 65; ++q) {
        float val = acc[q];
        for (int off = 16; off > 0; off >>= 1) val += __shfl_down(val, off, 32);
        if ((threadIdx.x & 31) == 0) atomicAdd(&mom[q], val);
    }
}

__device__ __forceinline__ float mom2(const float* mom, int a, int b)
{
    int lo = a < b ? a : b, hi = a < b ? b : a;
    return mom[10 + lo * 10 - (lo * (lo + 1)) / 2 + hi];
}

// conv1 pre-BN stats (analytic): h_c = b_c + sum_j w[c,j]*gf_j, gf = xyz[{3,4,5,0,1,2}]
__global__ void conv1_musig_kernel(const float* __restrict__ mom,
                                   const float* __restrict__ w64x6,
                                   const float* __restrict__ bias64,
                                   float* __restrict__ mu, float* __restrict__ rs)
{
    int c = threadIdx.x;
    if (c >= 64) return;
    const int g2x[6] = {3, 4, 5, 0, 1, 2};
    const float inv = 1.f / (float)BNK;
    const float* w = w64x6 + c * 6;
    float bias = bias64[c];
    float s1 = 0.f;
#pragma unroll
    for (int j = 0; j < 6; ++j) s1 += w[j] * mom[g2x[j]];
    float m = bias + s1 * inv;
    float e2 = 0.f;
#pragma unroll
    for (int i = 0; i < 6; ++i)
#pragma unroll
        for (int j = 0; j < 6; ++j)
            e2 += w[i] * w[j] * mom2(mom, g2x[i], g2x[j]);
    e2 = e2 * inv + 2.f * bias * (m - bias) + bias * bias;
    float var = e2 - m * m;
    mu[c] = m; rs[c] = rsqrtf(var + EPSBN);
}

// scorenet hidden pre-BN stats (analytic), 16 channels over the 10 xyz inputs
__global__ void score_musig_kernel(const float* __restrict__ mom,
                                   const float* __restrict__ w16x10,
                                   const float* __restrict__ bias16,
                                   float* __restrict__ mu, float* __restrict__ rs)
{
    int h = threadIdx.x;
    if (h >= 16) return;
    const float inv = 1.f / (float)BNK;
    const float* w = w16x10 + h * 10;
    float bias = bias16[h];
    float s1 = 0.f;
#pragma unroll
    for (int i = 0; i < 10; ++i) s1 += w[i] * mom[i];
    float m = bias + s1 * inv;
    float e2 = 0.f;
#pragma unroll
    for (int i = 0; i < 10; ++i)
#pragma unroll
        for (int j = 0; j < 10; ++j)
            e2 += w[i] * w[j] * mom2(mom, i, j);
    e2 = e2 * inv + 2.f * bias * (m - bias) + bias * bias;
    float var = e2 - m * m;
    mu[h] = m; rs[h] = rsqrtf(var + EPSBN);
}

// conv1 apply: BN + relu + max over K -> ft[bn,c] and xx[bn, c]
__global__ void conv1_apply_kernel(const float* __restrict__ xyz,
                                   const float* __restrict__ w, const float* __restrict__ bias,
                                   const float* __restrict__ gg, const float* __restrict__ be,
                                   const float* __restrict__ mu, const float* __restrict__ rs,
                                   float* __restrict__ ft, float* __restrict__ xx)
{
    __shared__ float s[KNB * 10];
    int bn = blockIdx.x;
    int t  = threadIdx.x;
    for (int i = t; i < KNB * 10; i += 64) s[i] = xyz[(size_t)bn * KNB * 10 + i];
    __syncthreads();
    int c = t;
    const float* wr = w + c * 6;
    float b0 = bias[c], gc = gg[c], bc = be[c], mc = mu[c], rc = rs[c];
    float mv = -3.4e38f;
    for (int k = 0; k < KNB; ++k) {
        const float* p = s + k * 10;
        float h = b0 + wr[0] * p[3] + wr[1] * p[4] + wr[2] * p[5]
                     + wr[3] * p[0] + wr[4] * p[1] + wr[5] * p[2];
        float v = fmaxf(gc * (h - mc) * rc + bc, 0.f);
        mv = fmaxf(mv, v);
    }
    ft[(size_t)bn * 64 + c]  = mv;
    xx[(size_t)bn * 320 + c] = mv;
}

// scorenet apply: h16 -> BN/relu -> w2 -> softmax over M -> score[bnk, m]
__global__ void score_apply_kernel(const float* __restrict__ xyz,
                                   const float* __restrict__ w1, const float* __restrict__ b1,
                                   const float* __restrict__ g1, const float* __restrict__ be1,
                                   const float* __restrict__ w2, const float* __restrict__ b2,
                                   const float* __restrict__ muS, const float* __restrict__ rsS,
                                   float* __restrict__ score)
{
    int p = blockIdx.x * blockDim.x + threadIdx.x;
    if (p >= BNK) return;
    float v[10];
    const float* src = xyz + (size_t)p * 10;
#pragma unroll
    for (int i = 0; i < 10; ++i) v[i] = src[i];
    float r[16];
#pragma unroll
    for (int h = 0; h < 16; ++h) {
        float hv = b1[h];
#pragma unroll
        for (int c = 0; c < 10; ++c) hv += w1[h * 10 + c] * v[c];
        hv = g1[h] * (hv - muS[h]) * rsS[h] + be1[h];
        r[h] = fmaxf(hv, 0.f);
    }
    float sv[MM]; float mx = -3.4e38f;
#pragma unroll
    for (int m = 0; m < MM; ++m) {
        float s = b2[m];
#pragma unroll
        for (int h = 0; h < 16; ++h) s += w2[m * 16 + h] * r[h];
        sv[m] = s; mx = fmaxf(mx, s);
    }
    float sum = 0.f;
#pragma unroll
    for (int m = 0; m < MM; ++m) { sv[m] = expf(sv[m] - mx); sum += sv[m]; }
    float is = 1.f / sum;
    float* o = score + (size_t)p * MM;
#pragma unroll
    for (int m = 0; m < MM; ++m) o[m] = sv[m] * is;
}

// assemble: out[bn,c] = sum_{k,m} sc*point[nbr, m*64+c] - sum_m ssum*center[bn, m*64+c]
// point = Cbuf + Pbuf, center = Cbuf
__global__ void assemble_kernel(const float* __restrict__ Cbuf, const float* __restrict__ Pbuf,
                                const float* __restrict__ score, const int* __restrict__ idxB,
                                float* __restrict__ outS)
{
    __shared__ float sc[KNB * MM];
    __shared__ float ssum[MM];
    __shared__ int   sid[KNB];
    int bn = blockIdx.x;
    int t  = threadIdx.x;
    int b  = bn / NN;
    for (int i = t; i < KNB * MM; i += 64) sc[i] = score[(size_t)bn * KNB * MM + i];
    if (t < KNB) sid[t] = idxB[bn * KNB + t];
    __syncthreads();
    if (t < MM) {
        float s = 0.f;
        for (int k = 0; k < KNB; ++k) s += sc[k * MM + t];
        ssum[t] = s;
    }
    __syncthreads();
    int c = t;
    float acc = 0.f;
    for (int k = 0; k < KNB; ++k) {
        size_t base = ((size_t)b * NN + sid[k]) * 512 + c;
        float w0 = sc[k * MM + 0], w1 = sc[k * MM + 1], w2 = sc[k * MM + 2], w3 = sc[k * MM + 3];
        float w4 = sc[k * MM + 4], w5 = sc[k * MM + 5], w6 = sc[k * MM + 6], w7 = sc[k * MM + 7];
        acc += w0 * (Cbuf[base +   0] + Pbuf[base +   0]);
        acc += w1 * (Cbuf[base +  64] + Pbuf[base +  64]);
        acc += w2 * (Cbuf[base + 128] + Pbuf[base + 128]);
        acc += w3 * (Cbuf[base + 192] + Pbuf[base + 192]);
        acc += w4 * (Cbuf[base + 256] + Pbuf[base + 256]);
        acc += w5 * (Cbuf[base + 320] + Pbuf[base + 320]);
        acc += w6 * (Cbuf[base + 384] + Pbuf[base + 384]);
        acc += w7 * (Cbuf[base + 448] + Pbuf[base + 448]);
    }
    size_t base0 = (size_t)bn * 512 + c;
#pragma unroll
    for (int m = 0; m < MM; ++m) acc -= ssum[m] * Cbuf[base0 + (size_t)m * 64];
    outS[(size_t)bn * 64 + c] = acc;
}

// per-output-channel batch stats of X (rows x C): one block per channel
__global__ void colstats_kernel(const float* __restrict__ X, int rows, int C,
                                float* __restrict__ mu, float* __restrict__ rs)
{
    int c = blockIdx.x, t = threadIdx.x;
    float s = 0.f, q = 0.f;
    for (int r = t; r < rows; r += blockDim.x) {
        float v = X[(size_t)r * C + c];
        s += v; q += v * v;
    }
    __shared__ float r1[256], r2[256];
    r1[t] = s; r2[t] = q;
    __syncthreads();
    for (int o = 128; o > 0; o >>= 1) {
        if (t < o) { r1[t] += r1[t + o]; r2[t] += r2[t + o]; }
        __syncthreads();
    }
    if (t == 0) {
        float m = r1[0] / (float)rows;
        float var = r2[0] / (float)rows - m * m;
        mu[c] = m; rs[c] = rsqrtf(var + EPSBN);
    }
}

__global__ void bnrelu_stage_kernel(const float* __restrict__ X,
                                    const float* __restrict__ g, const float* __restrict__ be,
                                    const float* __restrict__ mu, const float* __restrict__ rs,
                                    float* __restrict__ ft, float* __restrict__ xx, int chOff)
{
    int i = blockIdx.x * blockDim.x + threadIdx.x;
    if (i >= BN * 64) return;
    int c = i & 63, r = i >> 6;
    float v = fmaxf(g[c] * (X[i] - mu[c]) * rs[c] + be[c], 0.f);
    ft[i] = v;
    xx[(size_t)r * 320 + chOff + c] = v;
}

__global__ void bnrelu_inplace_kernel(float* __restrict__ X, int total, int C,
                                      const float* __restrict__ g, const float* __restrict__ be,
                                      const float* __restrict__ mu, const float* __restrict__ rs)
{
    int i = blockIdx.x * blockDim.x + threadIdx.x;
    if (i >= total) return;
    int c = i & (C - 1);   // C is a power of two here
    X[i] = fmaxf(g[c] * (X[i] - mu[c]) * rs[c] + be[c], 0.f);
}

__global__ void maxpool_kernel(const float* __restrict__ xc, float* __restrict__ g)
{
    int b = blockIdx.x >> 2;                        // 1024 ch / 256 threads = 4 chunks
    int c = (blockIdx.x & 3) * 256 + threadIdx.x;
    float m = -3.4e38f;
    for (int n = 0; n < NN; ++n) m = fmaxf(m, xc[((size_t)b * NN + n) * 1024 + c]);
    g[b * 1024 + c] = m;
}

__global__ void concat_kernel(const float* __restrict__ xx, const float* __restrict__ g,
                              float* __restrict__ h6in)
{
    int i = blockIdx.x * blockDim.x + threadIdx.x;
    const int total = BN * 1344;
    if (i >= total) return;
    int r = i / 1344, c = i - r * 1344;
    h6in[i] = (c < 320) ? xx[(size_t)r * 320 + c] : g[(r / NN) * 1024 + (c - 320)];
}

__global__ void final_kernel(const float* __restrict__ h9, const float* __restrict__ bias,
                             float* __restrict__ out)
{
    int i = blockIdx.x * blockDim.x + threadIdx.x;
    const int total = BB * 68 * NN;
    if (i >= total) return;
    int n = i % NN;
    int rem = i / NN;
    int o = rem % 68;
    int b = rem / 68;
    out[i] = h9[((size_t)b * NN + n) * 68 + o] + bias[o];
}

// ---------------------------------------------------------------------------
static inline void gemmf(const float* A, const float* W, float* Out,
                         int rows, int cols, int Ktot, int lda, int ldw, int ldo,
                         hipStream_t s)
{
    dim3 g(rows / 16, (cols + 63) / 64);
    gemm_wmma_kernel<<<g, 32, 0, s>>>(A, W, Out, Ktot, cols, lda, ldw, ldo);
}

extern "C" void kernel_launch(void* const* d_in, const int* in_sizes, int n_in,
                              void* d_out, int out_size, void* d_ws, size_t ws_size,
                              hipStream_t stream)
{
    (void)in_sizes; (void)n_in; (void)out_size; (void)ws_size;

    // ---- inputs (setup_inputs insertion order) ----
    const float* x        = (const float*)d_in[0];
    const float* conv1_w  = (const float*)d_in[1];
    const float* conv1_b  = (const float*)d_in[2];
    const float* conv1_g  = (const float*)d_in[3];
    const float* conv1_be = (const float*)d_in[4];
    const float *mats[4], *w1s[4], *b1s[4], *g1s[4], *be1s[4], *w2s[4], *b2s[4], *gs[4], *bes[4];
    for (int s = 0; s < 4; ++s) {
        int o = 5 + 9 * s;
        mats[s] = (const float*)d_in[o + 0];
        w1s[s]  = (const float*)d_in[o + 1];
        b1s[s]  = (const float*)d_in[o + 2];
        g1s[s]  = (const float*)d_in[o + 3];
        be1s[s] = (const float*)d_in[o + 4];
        w2s[s]  = (const float*)d_in[o + 5];
        b2s[s]  = (const float*)d_in[o + 6];
        gs[s]   = (const float*)d_in[o + 7];
        bes[s]  = (const float*)d_in[o + 8];
    }
    const float* convt_w = (const float*)d_in[41];
    const float* bnt_g   = (const float*)d_in[42];
    const float* bnt_be  = (const float*)d_in[43];
    const float* conv6_w = (const float*)d_in[44];
    const float* bn6_g   = (const float*)d_in[45];
    const float* bn6_be  = (const float*)d_in[46];
    const float* conv7_w = (const float*)d_in[47];
    const float* bn7_g   = (const float*)d_in[48];
    const float* bn7_be  = (const float*)d_in[49];
    const float* conv8_w = (const float*)d_in[50];
    const float* bn8_g   = (const float*)d_in[51];
    const float* bn8_be  = (const float*)d_in[52];
    const float* conv9_w = (const float*)d_in[53];
    const float* conv9_b = (const float*)d_in[54];

    // ---- workspace layout ----
    char* ws = (char*)d_ws;
    size_t cur = 0;
    auto alloc = [&](size_t bytes) -> void* {
        void* p = ws + cur;
        cur += (bytes + 255) & ~(size_t)255;
        return p;
    };
    int*   idxB     = (int*)  alloc(sizeof(int)   * BNK);
    float* xyz      = (float*)alloc(sizeof(float) * (size_t)BNK * 10);
    float* ft       = (float*)alloc(sizeof(float) * (size_t)BN * 64);
    float* xx       = (float*)alloc(sizeof(float) * (size_t)BN * 320);
    float* Cbuf     = (float*)alloc(sizeof(float) * (size_t)BN * 512);   // 16 MiB (256B-mult)
    float* Pbuf     = (float*)alloc(sizeof(float) * (size_t)BN * 512);   // contiguous after Cbuf
    float* xc       = Cbuf;   // (BN x 1024) aliases Cbuf..Pbuf after stages finish
    float* score    = (float*)alloc(sizeof(float) * (size_t)BNK * MM);
    float* stageOut = (float*)alloc(sizeof(float) * (size_t)BN * 64);
    float* matT     = (float*)alloc(sizeof(float) * 512 * 128);          // transposed `mat`
    float* mom      = (float*)alloc(sizeof(float) * 128);
    float* muA      = (float*)alloc(sizeof(float) * 1024);
    float* rsA      = (float*)alloc(sizeof(float) * 1024);
    float* muS      = (float*)alloc(sizeof(float) * 64);
    float* rsS      = (float*)alloc(sizeof(float) * 64);
    float* gpool    = (float*)alloc(sizeof(float) * BB * 1024);
    float* h6in     = (float*)alloc(sizeof(float) * (size_t)BN * 1344);
    float* h6       = (float*)alloc(sizeof(float) * (size_t)BN * 256);
    float* h7       = (float*)alloc(sizeof(float) * (size_t)BN * 256);
    float* h8       = (float*)alloc(sizeof(float) * (size_t)BN * 128);
    float* h9       = (float*)alloc(sizeof(float) * (size_t)BN * 68);

    // ---- geometry ----
    knn_kernel<<<BB * (NN / 256), 256, 0, stream>>>(x, idxB);
    xyz10_kernel<<<(BNK + 255) / 256, 256, 0, stream>>>(x, idxB, xyz);

    zero_kernel<<<1, 128, 0, stream>>>(mom, 128);
    moments_kernel<<<256, 256, 0, stream>>>(xyz, mom);

    // ---- conv1 + BN + relu + max over K ----
    conv1_musig_kernel<<<1, 64, 0, stream>>>(mom, conv1_w, conv1_b, muA, rsA);
    conv1_apply_kernel<<<BN, 64, 0, stream>>>(xyz, conv1_w, conv1_b, conv1_g, conv1_be,
                                              muA, rsA, ft, xx);

    // ---- 4 PAConv stages ----
    for (int s = 0; s < 4; ++s) {
        // mat (128 x 512, K-major) -> matT (512 x 128, out-major) for b128 loads
        transpose_kernel<<<(128 * 512 + 255) / 256, 256, 0, stream>>>(mats[s], matT, 128, 512);

        // center = ft @ mat[:64]   (matT rows 0..511, K slice 0..63)
        gemmf(ft, matT,      Cbuf, BN, 512, 64, 64, 128, 512, stream);
        // point = center + ft @ mat[64:]  (K slice 64..127)
        gemmf(ft, matT + 64, Pbuf, BN, 512, 64, 64, 128, 512, stream);

        score_musig_kernel<<<1, 16, 0, stream>>>(mom, w1s[s], b1s[s], muS, rsS);
        score_apply_kernel<<<(BNK + 255) / 256, 256, 0, stream>>>(
            xyz, w1s[s], b1s[s], g1s[s], be1s[s], w2s[s], b2s[s], muS, rsS, score);

        assemble_kernel<<<BN, 64, 0, stream>>>(Cbuf, Pbuf, score, idxB, stageOut);

        colstats_kernel<<<64, 256, 0, stream>>>(stageOut, BN, 64, muA, rsA);
        bnrelu_stage_kernel<<<(BN * 64 + 255) / 256, 256, 0, stream>>>(
            stageOut, gs[s], bes[s], muA, rsA, ft, xx, 64 * (s + 1));
    }

    // ---- convt (320 -> 1024) + BN + relu, global max pool ----
    gemmf(xx, convt_w, xc, BN, 1024, 320, 320, 320, 1024, stream);
    colstats_kernel<<<1024, 256, 0, stream>>>(xc, BN, 1024, muA, rsA);
    bnrelu_inplace_kernel<<<(BN * 1024 + 255) / 256, 256, 0, stream>>>(
        xc, BN * 1024, 1024, bnt_g, bnt_be, muA, rsA);
    maxpool_kernel<<<BB * 4, 256, 0, stream>>>(xc, gpool);

    // ---- concat [xx(320), broadcast g(1024)] -> 1344 ----
    concat_kernel<<<(BN * 1344 + 255) / 256, 256, 0, stream>>>(xx, gpool, h6in);

    // ---- conv6/7/8 with BN+relu ----
    gemmf(h6in, conv6_w, h6, BN, 256, 1344, 1344, 1344, 256, stream);
    colstats_kernel<<<256, 256, 0, stream>>>(h6, BN, 256, muA, rsA);
    bnrelu_inplace_kernel<<<(BN * 256 + 255) / 256, 256, 0, stream>>>(
        h6, BN * 256, 256, bn6_g, bn6_be, muA, rsA);

    gemmf(h6, conv7_w, h7, BN, 256, 256, 256, 256, 256, stream);
    colstats_kernel<<<256, 256, 0, stream>>>(h7, BN, 256, muA, rsA);
    bnrelu_inplace_kernel<<<(BN * 256 + 255) / 256, 256, 0, stream>>>(
        h7, BN * 256, 256, bn7_g, bn7_be, muA, rsA);

    gemmf(h7, conv8_w, h8, BN, 128, 256, 256, 256, 128, stream);
    colstats_kernel<<<128, 256, 0, stream>>>(h8, BN, 128, muA, rsA);
    bnrelu_inplace_kernel<<<(BN * 128 + 255) / 256, 256, 0, stream>>>(
        h8, BN * 128, 128, bn8_g, bn8_be, muA, rsA);

    // ---- conv9 (128 -> 68) + bias, transpose to (B,68,N) ----
    gemmf(h8, conv9_w, h9, BN, 68, 128, 128, 128, 68, stream);
    final_kernel<<<(BB * 68 * NN + 255) / 256, 256, 0, stream>>>(h9, conv9_b, (float*)d_out);
}
